// FullAttention_22686017257878
// MI455X (gfx1250) — matrix operand order repaired and checked by reading, not compiled, not run on hardware
//
#include <hip/hip_runtime.h>

// ---------------------------------------------------------------------------
// Full attention block for MI455X (gfx1250): bf16 WMMA everywhere, f32 accum.
// T=4096, C=1024, H=16, Dh=64.
// ---------------------------------------------------------------------------

typedef __bf16 bf16_t;
typedef __attribute__((ext_vector_type(16))) __bf16 v16bf;
typedef __attribute__((ext_vector_type(8)))  __bf16 v8bf;
typedef __attribute__((ext_vector_type(8)))  float   v8f;

#define SEQ    4096
#define DMODEL 1024
#define NH     16
#define DH     64

union Frag { v16bf f; v8bf h[2]; };

__device__ __forceinline__ v8f wmma_bf16(const Frag& a, const Frag& b, v8f c) {
  // D = A(16x32 bf16) x B(32x16 bf16) + C(16x16 f32)
  return __builtin_amdgcn_wmma_f32_16x16x32_bf16(
      /*neg_a=*/false, a.f, /*neg_b=*/false, b.f,
      /*c_mod=*/(short)0, c, /*reuse_a=*/false, /*reuse_b=*/false);
}

// ---------------------------------------------------------------------------
// 1) fp32 -> bf16 elementwise
// ---------------------------------------------------------------------------
__global__ void cvt_bf16_kernel(const float* __restrict__ in,
                                bf16_t* __restrict__ out, int n) {
  int i = blockIdx.x * blockDim.x + threadIdx.x;
  if (i < n) out[i] = (bf16_t)in[i];
}

// ---------------------------------------------------------------------------
// 2) transpose + convert: in [K][N] f32 -> out [N][K] bf16
// ---------------------------------------------------------------------------
__global__ void transpose_bf16_kernel(const float* __restrict__ in,
                                      bf16_t* __restrict__ out, int K, int N) {
  int i = blockIdx.x * blockDim.x + threadIdx.x;
  if (i >= K * N) return;
  int n = i % N, k = i / N;
  out[(size_t)n * K + k] = (bf16_t)in[i];
}

// ---------------------------------------------------------------------------
// 3) QKV GEMM: [SEQ x DMODEL] x [DMODEL x 3*DMODEL] (+bias), head-split
//    epilogue. Each wave: M=32 (2 frags) x N=64 (4 frags), K-loop step 32.
//    Q is pre-scaled by 1/sqrt(Dh)=0.125 (exact in bf16).
//    V is stored transposed [H][DH][SEQ] for the PV GEMM.
// ---------------------------------------------------------------------------
__global__ __launch_bounds__(128) void qkv_gemm_kernel(
    const bf16_t* __restrict__ X,    // [SEQ][DMODEL]
    const bf16_t* __restrict__ Wt,   // [3*DMODEL][DMODEL]  (W_qkv^T)
    const float*  __restrict__ bias, // [3*DMODEL]
    bf16_t* __restrict__ Qh,         // [NH][SEQ][DH]
    bf16_t* __restrict__ Kh,         // [NH][SEQ][DH]
    bf16_t* __restrict__ VTh) {      // [NH][DH][SEQ]
  const int wid  = blockIdx.x * (blockDim.x >> 5) + (threadIdx.x >> 5);
  const int lane = threadIdx.x & 31;
  const int l16  = lane & 15;
  const int hi   = lane >> 4;
  const int mtiles = SEQ / 32;                 // 128
  const int mt = wid % mtiles;
  const int nt = wid / mtiles;                 // < 48
  const int m0 = mt * 32;
  const int n0 = nt * 64;

  v8f acc[2][4];
#pragma unroll
  for (int mi = 0; mi < 2; ++mi)
#pragma unroll
    for (int ni = 0; ni < 4; ++ni) acc[mi][ni] = (v8f){};

  for (int k0 = 0; k0 < DMODEL; k0 += 32) {
    Frag a[2], b[4];
#pragma unroll
    for (int mi = 0; mi < 2; ++mi) {
      const bf16_t* p = X + (size_t)(m0 + mi * 16 + l16) * DMODEL + k0 + hi * 8;
      a[mi].h[0] = *(const v8bf*)(p);          // K = base+{0..7}
      a[mi].h[1] = *(const v8bf*)(p + 16);     // K = base+{16..23}
    }
#pragma unroll
    for (int ni = 0; ni < 4; ++ni) {
      const bf16_t* p = Wt + (size_t)(n0 + ni * 16 + l16) * DMODEL + k0 + hi * 8;
      b[ni].h[0] = *(const v8bf*)(p);          // K = half*8 + {0..7}
      b[ni].h[1] = *(const v8bf*)(p + 16);     // K = half*8 + {16..23}
    }
#pragma unroll
    for (int mi = 0; mi < 2; ++mi)
#pragma unroll
      for (int ni = 0; ni < 4; ++ni)
        acc[mi][ni] = wmma_bf16(a[mi], b[ni], acc[mi][ni]);
  }

#pragma unroll
  for (int mi = 0; mi < 2; ++mi)
#pragma unroll
    for (int ni = 0; ni < 4; ++ni)
#pragma unroll
      for (int j = 0; j < 8; ++j) {
        int t = m0 + mi * 16 + j + hi * 8;     // C/D layout: row j / j+8
        int n = n0 + ni * 16 + l16;            // col = lane&15
        float v = acc[mi][ni][j] + bias[n];
        if (n < DMODEL) {
          int h = n >> 6, d = n & 63;
          Qh[((size_t)(h * SEQ + t)) * DH + d] = (bf16_t)(v * 0.125f);
        } else if (n < 2 * DMODEL) {
          int n2 = n - DMODEL; int h = n2 >> 6, d = n2 & 63;
          Kh[((size_t)(h * SEQ + t)) * DH + d] = (bf16_t)v;
        } else {
          int n2 = n - 2 * DMODEL; int h = n2 >> 6, d = n2 & 63;
          VTh[((size_t)(h * DH + d)) * SEQ + t] = (bf16_t)v;
        }
      }
}

// ---------------------------------------------------------------------------
// 4) Flash attention, one wave per (head, 32-query block).
//    Computes S^T = K * Q^T so softmax reduces per column (= per query):
//    in-lane over 8 VGPRs + one shfl_xor(16). Online max/sum. P^T staged
//    through LDS to reshape into B-fragment layout for O^T = V^T * P^T.
//    32 queries/wave halves K/V L2 re-read traffic vs 16 and doubles WMMA
//    work per fragment load (16 WMMAs per 32-key tile vs 8 A-frag loads).
// ---------------------------------------------------------------------------
__global__ __launch_bounds__(128) void attn_kernel(
    const bf16_t* __restrict__ Qh,   // [NH][SEQ][DH], pre-scaled
    const bf16_t* __restrict__ Kh,   // [NH][SEQ][DH]
    const bf16_t* __restrict__ VTh,  // [NH][DH][SEQ]
    bf16_t* __restrict__ Ob) {       // [SEQ][DMODEL]
  __shared__ bf16_t Pl[4][32][40];   // per-wave [query][key] tile, padded row
  const int w    = threadIdx.x >> 5;
  const int wid  = blockIdx.x * 4 + w;
  const int lane = threadIdx.x & 31;
  const int l16  = lane & 15;
  const int hi   = lane >> 4;
  const int h    = wid >> 7;                   // wid / 128
  const int q0   = (wid & 127) * 32;

  // B-fragments of Q^T (resident): bq[qf][kk], column = query q0+qf*16+l16
  Frag bq[2][2];
#pragma unroll
  for (int qf = 0; qf < 2; ++qf) {
    const bf16_t* p = Qh + ((size_t)(h * SEQ + q0 + qf * 16 + l16)) * DH;
#pragma unroll
    for (int kk = 0; kk < 2; ++kk) {
      bq[qf][kk].h[0] = *(const v8bf*)(p + kk * 32 + hi * 8);
      bq[qf][kk].h[1] = *(const v8bf*)(p + kk * 32 + hi * 8 + 16);
    }
  }

  v8f acc[4][2];                               // O^T: 64 d-rows x 32 queries
#pragma unroll
  for (int df = 0; df < 4; ++df)
#pragma unroll
    for (int qf = 0; qf < 2; ++qf) acc[df][qf] = (v8f){};
  float m_run[2] = {-1e30f, -1e30f};
  float l_run[2] = {0.0f, 0.0f};

  for (int kt = 0; kt < SEQ / 32; ++kt) {
    // ---- scores S^T: 32 keys x 32 queries, s[kh][qf] frags of 16x16 ----
    Frag ak[2][2];
#pragma unroll
    for (int kh = 0; kh < 2; ++kh) {
      const bf16_t* p =
          Kh + ((size_t)(h * SEQ + kt * 32 + kh * 16 + l16)) * DH + hi * 8;
      ak[kh][0].h[0] = *(const v8bf*)(p);      // Dh {0..7}+kbase
      ak[kh][0].h[1] = *(const v8bf*)(p + 16); // Dh {16..23}+kbase
      ak[kh][1].h[0] = *(const v8bf*)(p + 32); // Dh chunk 1
      ak[kh][1].h[1] = *(const v8bf*)(p + 48);
    }
    v8f s[2][2];
#pragma unroll
    for (int kh = 0; kh < 2; ++kh)
#pragma unroll
      for (int qf = 0; qf < 2; ++qf) {
        v8f z = (v8f){};
        z = wmma_bf16(ak[kh][0], bq[qf][0], z);
        z = wmma_bf16(ak[kh][1], bq[qf][1], z);
        s[kh][qf] = z;
      }

    // ---- online softmax per query column (per qf fragment) ----
    float corr[2];
#pragma unroll
    for (int qf = 0; qf < 2; ++qf) {
      float pm = -1e30f;
#pragma unroll
      for (int kh = 0; kh < 2; ++kh)
#pragma unroll
        for (int j = 0; j < 8; ++j) pm = fmaxf(pm, s[kh][qf][j]);
      pm = fmaxf(pm, __shfl_xor(pm, 16, 32));  // combine lane halves
      float new_m = fmaxf(m_run[qf], pm);
      corr[qf] = __expf(m_run[qf] - new_m);
      float psum = 0.0f;
#pragma unroll
      for (int kh = 0; kh < 2; ++kh)
#pragma unroll
        for (int j = 0; j < 8; ++j) {
          float pv = __expf(s[kh][qf][j] - new_m);
          psum += pv;
          Pl[w][qf * 16 + l16][kh * 16 + hi * 8 + j] = (bf16_t)pv;
        }
      psum += __shfl_xor(psum, 16, 32);
      l_run[qf] = l_run[qf] * corr[qf] + psum;
      m_run[qf] = new_m;
    }
#pragma unroll
    for (int df = 0; df < 4; ++df)
#pragma unroll
      for (int qf = 0; qf < 2; ++qf)
#pragma unroll
        for (int j = 0; j < 8; ++j) acc[df][qf][j] *= corr[qf];

    // same-wave LDS RAW: DS ops are in-order per wave; fence the compiler
    // and drain DScnt before the reshaped reload.
    asm volatile("s_wait_dscnt 0x0" ::: "memory");

    // B fragments of P^T: column = query qf*16+l16, K = keys hi*8..+7, 16+hi*8..+7
    Frag bp[2];
#pragma unroll
    for (int qf = 0; qf < 2; ++qf) {
      bp[qf].h[0] = *(const v8bf*)(&Pl[w][qf * 16 + l16][hi * 8]);
      bp[qf].h[1] = *(const v8bf*)(&Pl[w][qf * 16 + l16][hi * 8 + 16]);
    }

    // ---- O^T += V^T(16 d x 32 keys) * P^T(32 keys x 16 q) per frag ----
#pragma unroll
    for (int df = 0; df < 4; ++df) {
      Frag av;
      const bf16_t* p =
          VTh + ((size_t)(h * DH + df * 16 + l16)) * SEQ + kt * 32 + hi * 8;
      av.h[0] = *(const v8bf*)(p);
      av.h[1] = *(const v8bf*)(p + 16);
#pragma unroll
      for (int qf = 0; qf < 2; ++qf)
        acc[df][qf] = wmma_bf16(av, bp[qf], acc[df][qf]);
    }
  }

#pragma unroll
  for (int qf = 0; qf < 2; ++qf) {
    const float inv = 1.0f / l_run[qf];
    const int q = q0 + qf * 16 + l16;
#pragma unroll
    for (int df = 0; df < 4; ++df)
#pragma unroll
      for (int j = 0; j < 8; ++j) {
        int d = df * 16 + j + hi * 8;
        Ob[(size_t)q * DMODEL + h * DH + d] = (bf16_t)(acc[df][qf][j] * inv);
      }
  }
}

// ---------------------------------------------------------------------------
// 5) Output projection: [SEQ x DMODEL] x [DMODEL x DMODEL] + bias -> fp32
// ---------------------------------------------------------------------------
__global__ __launch_bounds__(128) void out_gemm_kernel(
    const bf16_t* __restrict__ A,    // [SEQ][DMODEL] attn output (bf16)
    const bf16_t* __restrict__ Wt,   // [DMODEL][DMODEL]  (W_out^T)
    const float*  __restrict__ bias, // [DMODEL]
    float* __restrict__ out) {       // [SEQ][DMODEL]
  const int wid  = blockIdx.x * (blockDim.x >> 5) + (threadIdx.x >> 5);
  const int lane = threadIdx.x & 31;
  const int l16  = lane & 15;
  const int hi   = lane >> 4;
  const int mtiles = SEQ / 32;                 // 128
  const int mt = wid % mtiles;
  const int nt = wid / mtiles;                 // < 16
  const int m0 = mt * 32;
  const int n0 = nt * 64;

  v8f acc[2][4];
#pragma unroll
  for (int mi = 0; mi < 2; ++mi)
#pragma unroll
    for (int ni = 0; ni < 4; ++ni) acc[mi][ni] = (v8f){};

  for (int k0 = 0; k0 < DMODEL; k0 += 32) {
    Frag a[2], b[4];
#pragma unroll
    for (int mi = 0; mi < 2; ++mi) {
      const bf16_t* p = A + (size_t)(m0 + mi * 16 + l16) * DMODEL + k0 + hi * 8;
      a[mi].h[0] = *(const v8bf*)(p);
      a[mi].h[1] = *(const v8bf*)(p + 16);
    }
#pragma unroll
    for (int ni = 0; ni < 4; ++ni) {
      const bf16_t* p = Wt + (size_t)(n0 + ni * 16 + l16) * DMODEL + k0 + hi * 8;
      b[ni].h[0] = *(const v8bf*)(p);
      b[ni].h[1] = *(const v8bf*)(p + 16);
    }
#pragma unroll
    for (int mi = 0; mi < 2; ++mi)
#pragma unroll
      for (int ni = 0; ni < 4; ++ni)
        acc[mi][ni] = wmma_bf16(a[mi], b[ni], acc[mi][ni]);
  }

#pragma unroll
  for (int mi = 0; mi < 2; ++mi)
#pragma unroll
    for (int ni = 0; ni < 4; ++ni)
#pragma unroll
      for (int j = 0; j < 8; ++j) {
        int t = m0 + mi * 16 + j + hi * 8;
        int n = n0 + ni * 16 + l16;
        out[(size_t)t * DMODEL + n] = acc[mi][ni][j] + bias[n];
      }
}

// ---------------------------------------------------------------------------
extern "C" void kernel_launch(void* const* d_in, const int* in_sizes, int n_in,
                              void* d_out, int out_size, void* d_ws,
                              size_t ws_size, hipStream_t stream) {
  const float* x     = (const float*)d_in[0];   // [1,4096,1024]
  const float* W_qkv = (const float*)d_in[1];   // [1024,3072]
  const float* b_qkv = (const float*)d_in[2];   // [3072]
  const float* W_out = (const float*)d_in[3];   // [1024,1024]
  const float* b_out = (const float*)d_in[4];   // [1024]
  float* out = (float*)d_out;

  // Workspace carve-up (all bf16): 24M elements = 48 MB total.
  bf16_t* Xb    = (bf16_t*)d_ws;                        // SEQ*DMODEL
  bf16_t* WqkvT = Xb    + (size_t)SEQ * DMODEL;         // 3*DMODEL*DMODEL
  bf16_t* WoutT = WqkvT + (size_t)3 * DMODEL * DMODEL;  // DMODEL*DMODEL
  bf16_t* Qh    = WoutT + (size_t)DMODEL * DMODEL;      // SEQ*DMODEL
  bf16_t* Kh    = Qh    + (size_t)SEQ * DMODEL;
  bf16_t* VTh   = Kh    + (size_t)SEQ * DMODEL;
  bf16_t* Ob    = VTh   + (size_t)SEQ * DMODEL;

  const int nX = SEQ * DMODEL;
  cvt_bf16_kernel<<<(nX + 255) / 256, 256, 0, stream>>>(x, Xb, nX);

  const int nWqkv = DMODEL * 3 * DMODEL;
  transpose_bf16_kernel<<<(nWqkv + 255) / 256, 256, 0, stream>>>(
      W_qkv, WqkvT, DMODEL, 3 * DMODEL);
  const int nWout = DMODEL * DMODEL;
  transpose_bf16_kernel<<<(nWout + 255) / 256, 256, 0, stream>>>(
      W_out, WoutT, DMODEL, DMODEL);

  // QKV GEMM: (SEQ/32)*(3*DMODEL/64) = 128*48 = 6144 waves, 4 waves/block
  qkv_gemm_kernel<<<6144 / 4, 128, 0, stream>>>(Xb, WqkvT, b_qkv, Qh, Kh, VTh);

  // Attention: NH*(SEQ/32) = 2048 waves
  attn_kernel<<<2048 / 4, 128, 0, stream>>>(Qh, Kh, VTh, Ob);

  // Output projection: (SEQ/32)*(DMODEL/64) = 2048 waves
  out_gemm_kernel<<<2048 / 4, 128, 0, stream>>>(Ob, WoutT, b_out, out);
}
